// DeepHGNNP_88295937671287
// MI455X (gfx1250) — compile-verified, matrix operand order checked by reading, not run on hardware
//
#include <hip/hip_runtime.h>
#include <hip/hip_bf16.h>

// ---------------------------------------------------------------------------
// DeepHGNN+ forward on MI455X (gfx1250, wave32, WMMA).
//  - GEMMs via v_wmma_f32_16x16x32_bf16 (bf16 in, fp32 accumulate)
//  - B fragments preloaded so the 8 WMMAs issue back-to-back (no per-WMMA
//    s_wait_dscnt 0 stall)
//  - segment sums via L2-resident float atomics + global_prefetch ahead
//  - LN+ReLU fused into the per-layer theta GEMM (A-tile built in LDS)
// ---------------------------------------------------------------------------

typedef __attribute__((ext_vector_type(16))) __bf16 v16bf;
typedef __attribute__((ext_vector_type(8)))  float  v8f;

#define N_NODES 100000
#define M_EDGES 20000
#define P_PAIRS 1600000
#define C_IN    768
#define HID     128
#define OUT_C   16
#define EPS_LN  1e-5f

// float -> bf16, round to nearest even
static __device__ __forceinline__ unsigned short f2bf(float f) {
  unsigned int u = __float_as_uint(f);
  unsigned int r = u + 0x7FFFu + ((u >> 16) & 1u);
  return (unsigned short)(r >> 16);
}

union FragU { v16bf v; unsigned int u[8]; };

// Build a 16x32 bf16 WMMA operand fragment from an LDS tile stored row-major
// as bf16 (A: [m][k], B staged transposed as [n][k]).  Matches the CDNA5
// 16-bit A-matrix layout: lane L holds row m=L&15; half kh=(L>>4)*8;
// VGPR j packs K = (j>>2)*16 + kh + (j&3)*2 and K+1.
static __device__ __forceinline__ v16bf ld_frag(const unsigned short* base,
                                                int stride, int rowBase,
                                                int lane, int k0) {
  FragU f;
  const int m  = lane & 15;
  const int kh = (lane >> 4) << 3;
  const unsigned short* p = base + (rowBase + m) * stride + k0 + kh;
#pragma unroll
  for (int j = 0; j < 8; ++j) {
    const int kk = ((j >> 2) << 4) + ((j & 3) << 1);
    f.u[j] = *(const unsigned int*)(p + kk);   // aligned: all terms even
  }
  return f.v;
}

// ---------------------------------------------------------------------------
// Encoder: x = X @ W_enc + b_enc.   Block = 128 rows x 128 cols, K = 768.
// ---------------------------------------------------------------------------
__global__ __launch_bounds__(256) void encoder_kernel(
    const float* __restrict__ X, const unsigned short* __restrict__ Wb,
    const float* __restrict__ b_enc, float* __restrict__ xo, int n) {
  __shared__ unsigned short aLds[128 * 36];   // 128 rows x 32 K (bf16), pad 4
  __shared__ unsigned short wLds[128 * 36];   // transposed: 128 n x 32 K
  const int tid = threadIdx.x;
  const int wv = tid >> 5, lane = tid & 31;
  const int rowBase = blockIdx.x * 128;

  v8f acc[8];
#pragma unroll
  for (int t = 0; t < 8; ++t) acc[t] = v8f{};

  for (int kc = 0; kc < C_IN; kc += 32) {
    // stage A (f32 -> bf16), coalesced rows of 32 floats
#pragma unroll
    for (int i = tid; i < 128 * 32; i += 256) {
      const int r = i >> 5, c = i & 31;
      const int row = rowBase + r;
      const float v = (row < n) ? X[(size_t)row * C_IN + kc + c] : 0.0f;
      aLds[r * 36 + c] = f2bf(v);
    }
    // stage W chunk transposed: wLds[n][k]
#pragma unroll
    for (int i = tid; i < 32 * 128; i += 256) {
      const int k = i >> 7, c = i & 127;
      wLds[c * 36 + k] = Wb[(kc + k) * HID + c];
    }
    __syncthreads();
    // preload all fragments, then chain the 8 independent WMMAs
    const v16bf a = ld_frag(aLds, 36, wv * 16, lane, 0);
    v16bf bfr[8];
#pragma unroll
    for (int t = 0; t < 8; ++t) bfr[t] = ld_frag(wLds, 36, t * 16, lane, 0);
#pragma unroll
    for (int t = 0; t < 8; ++t) {
      acc[t] = __builtin_amdgcn_wmma_f32_16x16x32_bf16(
          false, a, false, bfr[t], (short)0, acc[t], false, false);
    }
    __syncthreads();
  }

  // C/D layout: element (M = r + 8*(lane>>4), N = lane&15) in VGPR r
  const int mo = (lane >> 4) << 3, nn = lane & 15;
#pragma unroll
  for (int t = 0; t < 8; ++t) {
    const float bias = b_enc[t * 16 + nn];
#pragma unroll
    for (int r = 0; r < 8; ++r) {
      const int row = rowBase + wv * 16 + mo + r;
      if (row < n) xo[(size_t)row * HID + t * 16 + nn] = acc[t][r] + bias;
    }
  }
}

// ---------------------------------------------------------------------------
// Per-layer theta: h = relu(layernorm(x; g,b)) @ Wt + bt    (K = HID = 128)
// LN+ReLU computed directly into the bf16 A-tile in LDS.
// ---------------------------------------------------------------------------
__global__ __launch_bounds__(256) void theta_kernel(
    const float* __restrict__ x, const float* __restrict__ g,
    const float* __restrict__ b, const unsigned short* __restrict__ Wb,
    const float* __restrict__ bt, float* __restrict__ h, int n) {
  __shared__ unsigned short aLds[128 * 136];  // 128 rows x 128 K bf16, pad 8
  __shared__ unsigned short wLds[128 * 68];   // transposed 128 n x 64 K chunk
  const int tid = threadIdx.x, wv = tid >> 5, lane = tid & 31;
  const int rowBase = blockIdx.x * 128;

  // Phase 1: LN + ReLU -> bf16 A tile (one row per thread, threads 0..127)
  if (tid < 128) {
    const int row = rowBase + tid;
    if (row < n) {
      const float4* xr = (const float4*)(x + (size_t)row * HID);
      float s = 0.f, ss = 0.f;
#pragma unroll
      for (int i = 0; i < 32; ++i) {
        const float4 v = xr[i];
        s  += v.x + v.y + v.z + v.w;
        ss += v.x * v.x + v.y * v.y + v.z * v.z + v.w * v.w;
      }
      const float mu  = s * (1.0f / HID);
      const float var = ss * (1.0f / HID) - mu * mu;
      const float rs  = rsqrtf(var + EPS_LN);
#pragma unroll
      for (int i = 0; i < 32; ++i) {
        const float4 v  = xr[i];
        const float4 gg = ((const float4*)g)[i];
        const float4 bb = ((const float4*)b)[i];
        aLds[tid * 136 + 4 * i + 0] = f2bf(fmaxf((v.x - mu) * rs * gg.x + bb.x, 0.f));
        aLds[tid * 136 + 4 * i + 1] = f2bf(fmaxf((v.y - mu) * rs * gg.y + bb.y, 0.f));
        aLds[tid * 136 + 4 * i + 2] = f2bf(fmaxf((v.z - mu) * rs * gg.z + bb.z, 0.f));
        aLds[tid * 136 + 4 * i + 3] = f2bf(fmaxf((v.w - mu) * rs * gg.w + bb.w, 0.f));
      }
    } else {
#pragma unroll
      for (int k = 0; k < HID; ++k) aLds[tid * 136 + k] = 0;
    }
  }

  v8f acc[8];
#pragma unroll
  for (int t = 0; t < 8; ++t) acc[t] = v8f{};

  for (int kc = 0; kc < HID; kc += 64) {
    __syncthreads();  // A ready / previous chunk consumed
#pragma unroll
    for (int i = tid; i < 64 * 128; i += 256) {
      const int k = i >> 7, c = i & 127;
      wLds[c * 68 + k] = Wb[(kc + k) * HID + c];
    }
    __syncthreads();
#pragma unroll
    for (int k0 = 0; k0 < 64; k0 += 32) {
      const v16bf a = ld_frag(aLds, 136, wv * 16, lane, kc + k0);
      v16bf bfr[8];
#pragma unroll
      for (int t = 0; t < 8; ++t) bfr[t] = ld_frag(wLds, 68, t * 16, lane, k0);
#pragma unroll
      for (int t = 0; t < 8; ++t) {
        acc[t] = __builtin_amdgcn_wmma_f32_16x16x32_bf16(
            false, a, false, bfr[t], (short)0, acc[t], false, false);
      }
    }
  }

  const int mo = (lane >> 4) << 3, nn = lane & 15;
#pragma unroll
  for (int t = 0; t < 8; ++t) {
    const float bias = bt[t * 16 + nn];
#pragma unroll
    for (int r = 0; r < 8; ++r) {
      const int row = rowBase + wv * 16 + mo + r;
      if (row < n) h[(size_t)row * HID + t * 16 + nn] = acc[t][r] + bias;
    }
  }
}

// ---------------------------------------------------------------------------
// Segment add: dst[idxB[p]] += src[idxA[p]]  (one thread per pair x 4 chans)
// Prefetch the gather row ~64 pairs ahead (global_prefetch_b8 path).
// ---------------------------------------------------------------------------
__global__ void seg_add_kernel(const float* __restrict__ src,
                               const int* __restrict__ idxA,
                               const int* __restrict__ idxB,
                               float* __restrict__ dst, int P) {
  const long long gid = (long long)blockIdx.x * blockDim.x + threadIdx.x;
  if (gid >= (long long)P * 32) return;
  const int p = (int)(gid >> 5);
  const int c = ((int)gid & 31) << 2;
  if (p + 64 < P) {
    __builtin_prefetch(src + (size_t)idxA[p + 64] * HID + c, 0, 3);
  }
  const int a = idxA[p], d = idxB[p];
  const float4 val = *(const float4*)(src + (size_t)a * HID + c);
  float* q = dst + (size_t)d * HID + c;
  atomicAdd(q + 0, val.x);
  atomicAdd(q + 1, val.y);
  atomicAdd(q + 2, val.z);
  atomicAdd(q + 3, val.w);
}

__global__ void count_kernel(const int* __restrict__ vi,
                             const int* __restrict__ ei,
                             float* __restrict__ cv, float* __restrict__ ce,
                             int P) {
  const int gid = blockIdx.x * blockDim.x + threadIdx.x;
  if (gid >= P) return;
  atomicAdd(&cv[vi[gid]], 1.0f);
  atomicAdd(&ce[ei[gid]], 1.0f);
}

__global__ void inv_kernel(float* __restrict__ c, int n) {
  const int gid = blockIdx.x * blockDim.x + threadIdx.x;
  if (gid < n) c[gid] = 1.0f / fmaxf(c[gid], 1.0f);
}

__global__ void cvt_bf16_kernel(const float* __restrict__ src,
                                unsigned short* __restrict__ dst, int n) {
  const int gid = blockIdx.x * blockDim.x + threadIdx.x;
  if (gid < n) dst[gid] = f2bf(src[gid]);
}

__global__ void scale_rows_kernel(float* __restrict__ A,
                                  const float* __restrict__ s,
                                  long long total) {
  const long long gid = (long long)blockIdx.x * blockDim.x + threadIdx.x;
  if (gid < total) A[gid] *= s[gid >> 7];   // HID == 128
}

// x += relu(h * inv_cv[node])
__global__ void finish_kernel(float* __restrict__ x,
                              const float* __restrict__ h,
                              const float* __restrict__ cv, long long total) {
  const long long gid = (long long)blockIdx.x * blockDim.x + threadIdx.x;
  if (gid < total) x[gid] += fmaxf(h[gid] * cv[gid >> 7], 0.0f);
}

// ---------------------------------------------------------------------------
// Head: out = log_softmax(x @ W_out + b_out)
// ---------------------------------------------------------------------------
__global__ __launch_bounds__(256) void head_kernel(
    const float* __restrict__ x, const float* __restrict__ Wo,
    const float* __restrict__ bo, float* __restrict__ out, int n) {
  __shared__ float wo[HID * OUT_C];
  __shared__ float bos[OUT_C];
  for (int i = threadIdx.x; i < HID * OUT_C; i += 256) wo[i] = Wo[i];
  if (threadIdx.x < OUT_C) bos[threadIdx.x] = bo[threadIdx.x];
  __syncthreads();
  const int row = blockIdx.x * 256 + threadIdx.x;
  if (row >= n) return;

  float z[OUT_C];
#pragma unroll
  for (int j = 0; j < OUT_C; ++j) z[j] = bos[j];
  const float4* xr = (const float4*)(x + (size_t)row * HID);
#pragma unroll
  for (int i = 0; i < 32; ++i) {
    const float4 v = xr[i];
#pragma unroll
    for (int j = 0; j < OUT_C; ++j) {
      z[j] += v.x * wo[(4 * i + 0) * OUT_C + j] +
              v.y * wo[(4 * i + 1) * OUT_C + j] +
              v.z * wo[(4 * i + 2) * OUT_C + j] +
              v.w * wo[(4 * i + 3) * OUT_C + j];
    }
  }
  float m = z[0];
#pragma unroll
  for (int j = 1; j < OUT_C; ++j) m = fmaxf(m, z[j]);
  float s = 0.f;
#pragma unroll
  for (int j = 0; j < OUT_C; ++j) s += __expf(z[j] - m);
  const float l = __logf(s);
#pragma unroll
  for (int j = 0; j < OUT_C; ++j) out[(size_t)row * OUT_C + j] = z[j] - m - l;
}

// ---------------------------------------------------------------------------
extern "C" void kernel_launch(void* const* d_in, const int* in_sizes, int n_in,
                              void* d_out, int out_size, void* d_ws,
                              size_t ws_size, hipStream_t stream) {
  (void)in_sizes; (void)n_in; (void)out_size; (void)ws_size;
  const float* X     = (const float*)d_in[0];
  const int*   v_idx = (const int*)d_in[1];
  const int*   e_idx = (const int*)d_in[2];
  const float* W_enc = (const float*)d_in[3];
  const float* b_enc = (const float*)d_in[4];
  const float* ln_g  = (const float*)d_in[5];
  const float* ln_b  = (const float*)d_in[6];
  const float* Wt    = (const float*)d_in[7];
  const float* bt    = (const float*)d_in[8];
  const float* W_out = (const float*)d_in[9];
  const float* b_out = (const float*)d_in[10];
  float* out = (float*)d_out;

  unsigned char* ws = (unsigned char*)d_ws;
  size_t off = 0;
  auto take = [&](size_t bytes) -> void* {
    void* p = ws + off;
    off += (bytes + 255) & ~(size_t)255;
    return p;
  };
  float* x   = (float*)take((size_t)N_NODES * HID * 4);
  float* h   = (float*)take((size_t)N_NODES * HID * 4);
  float* Xe  = (float*)take((size_t)M_EDGES * HID * 4);
  float* icv = (float*)take((size_t)N_NODES * 4);
  float* ice = (float*)take((size_t)M_EDGES * 4);
  unsigned short* Wenc_b = (unsigned short*)take((size_t)C_IN * HID * 2);
  unsigned short* Wt_b   = (unsigned short*)take((size_t)OUT_C * HID * HID * 2);

  // degree normalizers
  hipMemsetAsync(icv, 0, (size_t)N_NODES * 4, stream);
  hipMemsetAsync(ice, 0, (size_t)M_EDGES * 4, stream);
  count_kernel<<<(P_PAIRS + 255) / 256, 256, 0, stream>>>(v_idx, e_idx, icv, ice, P_PAIRS);
  inv_kernel<<<(N_NODES + 255) / 256, 256, 0, stream>>>(icv, N_NODES);
  inv_kernel<<<(M_EDGES + 255) / 256, 256, 0, stream>>>(ice, M_EDGES);

  // bf16 weight copies
  cvt_bf16_kernel<<<(C_IN * HID + 255) / 256, 256, 0, stream>>>(W_enc, Wenc_b, C_IN * HID);
  cvt_bf16_kernel<<<(OUT_C * HID * HID + 255) / 256, 256, 0, stream>>>(Wt, Wt_b, OUT_C * HID * HID);

  // encoder
  encoder_kernel<<<(N_NODES + 127) / 128, 256, 0, stream>>>(X, Wenc_b, b_enc, x, N_NODES);

  // 16 HGNN+ layers
  const int segBlocks = (int)(((long long)P_PAIRS * 32 + 255) / 256);
  for (int l = 0; l < OUT_C; ++l) {
    theta_kernel<<<(N_NODES + 127) / 128, 256, 0, stream>>>(
        x, ln_g + l * HID, ln_b + l * HID, Wt_b + (size_t)l * HID * HID,
        bt + l * HID, h, N_NODES);
    hipMemsetAsync(Xe, 0, (size_t)M_EDGES * HID * 4, stream);
    seg_add_kernel<<<segBlocks, 256, 0, stream>>>(h, v_idx, e_idx, Xe, P_PAIRS);
    scale_rows_kernel<<<(M_EDGES * HID) / 256, 256, 0, stream>>>(
        Xe, ice, (long long)M_EDGES * HID);
    hipMemsetAsync(h, 0, (size_t)N_NODES * HID * 4, stream);
    seg_add_kernel<<<segBlocks, 256, 0, stream>>>(Xe, e_idx, v_idx, h, P_PAIRS);
    finish_kernel<<<(N_NODES * HID + 255) / 256, 256, 0, stream>>>(
        x, h, icv, (long long)N_NODES * HID);
  }

  head_kernel<<<(N_NODES + 255) / 256, 256, 0, stream>>>(x, W_out, b_out, out, N_NODES);
}
